// GatedConditionalCausalAttention_64587718197306
// MI455X (gfx1250) — matrix operand order repaired and checked by reading, not compiled, not run on hardware
//
#include <hip/hip_runtime.h>
#include <hip/hip_bf16.h>
#include <stdint.h>

typedef __attribute__((ext_vector_type(16))) __bf16 v16bf;
typedef __attribute__((ext_vector_type(8)))  __bf16 v8bf;
typedef __attribute__((ext_vector_type(8)))  float  v8f;

constexpr int Bsz  = 256;
constexpr int Tsz  = 87;
constexpr int Csz  = 1024;
constexpr int Hn   = 16;
constexpr int Dd   = 64;
constexpr int BLKm = 29;
constexpr int Mrows = Bsz * Tsz;   // 22272 = 174 * 128

// Issue one 16-byte async global->LDS copy (per-lane). GVS addressing:
// vdst = LDS byte-offset VGPR, vaddr = 32-bit byte offset, saddr = SGPR base.
#define ASYNC_B128(ldsoff, goff, gbase)                                     \
  asm volatile("global_load_async_to_lds_b128 %0, %1, %2"                   \
               :: "v"(ldsoff), "v"(goff), "s"(gbase) : "memory")

#define WAIT_ASYNC0() asm volatile("s_wait_asynccnt 0x0" ::: "memory")

// ---------------- weight transpose + f32->bf16: out[n][k] = (bf16)in[k][n] ----
__global__ __launch_bounds__(256)
void transpose_cvt_kernel(const float* __restrict__ in, __bf16* __restrict__ out) {
  __shared__ float t[32][33];
  const int bx = blockIdx.x;          // n tile
  const int by = blockIdx.y;          // k tile
  const int lx = threadIdx.x & 31;
  const int ly = threadIdx.x >> 5;    // 0..7
#pragma unroll
  for (int r = 0; r < 32; r += 8)
    t[ly + r][lx] = in[(size_t)(by * 32 + ly + r) * Csz + bx * 32 + lx];
  __syncthreads();
#pragma unroll
  for (int r = 0; r < 32; r += 8)
    out[(size_t)(bx * 32 + ly + r) * Csz + by * 32 + lx] = (__bf16)t[lx][ly + r];
}

// ---------------- LayerNorm: one row (C=1024) per block ----------------
__global__ __launch_bounds__(256)
void layernorm_kernel(const float* __restrict__ x,
                      const float* __restrict__ gamma,
                      const float* __restrict__ beta,
                      __bf16* __restrict__ xn) {
  const int row = blockIdx.x;
  const int tid = threadIdx.x;
  const float* xr = x + (size_t)row * Csz;
  float v[4];
  float s = 0.f, s2 = 0.f;
#pragma unroll
  for (int i = 0; i < 4; ++i) {
    float t = xr[tid + i * 256];
    v[i] = t; s += t; s2 += t * t;
  }
  __shared__ float sh[256];
  sh[tid] = s; __syncthreads();
  for (int off = 128; off > 0; off >>= 1) { if (tid < off) sh[tid] += sh[tid + off]; __syncthreads(); }
  const float mean = sh[0] * (1.f / Csz);
  __syncthreads();
  sh[tid] = s2; __syncthreads();
  for (int off = 128; off > 0; off >>= 1) { if (tid < off) sh[tid] += sh[tid + off]; __syncthreads(); }
  const float var  = sh[0] * (1.f / Csz) - mean * mean;
  const float rstd = rsqrtf(var + 1e-5f);
  __bf16* xo = xn + (size_t)row * Csz;
#pragma unroll
  for (int i = 0; i < 4; ++i) {
    const int c = tid + i * 256;
    xo[c] = (__bf16)(((v[i] - mean) * rstd) * gamma[c] + beta[c]);
  }
}

// ---------------- WMMA GEMM: [M,1024] x Wt[1024,1024] (Wt = W^T) ----------------
// 128x128 block tile, 8 waves (4 M x 2 N), each wave 32x64 via 2x4 wmma tiles.
// Double-buffered LDS filled by global_load_async_to_lds_b128 (ASYNCcnt).
#define EP_SILU 0
#define EP_BIAS 1
#define EP_GATE 2
#define EP_OUTF 3

template <int MODE>
__global__ __launch_bounds__(256)
void gemm_wmma(const __bf16* __restrict__ A,    // [M, 1024] bf16
               const __bf16* __restrict__ Wt,   // [1024, 1024] bf16 = W^T, row-major [n][k]
               const float* __restrict__ bias,  // [1024]
               const __bf16* __restrict__ gate, // [M, 1024] bf16 (EP_GATE) or null
               void* __restrict__ outv) {
  constexpr int K = 1024, N = 1024;
  constexpr int LDA = 40;                 // halfs; 80B pitch -> conflict-free 16B lane reads
  constexpr int LDB = 40;
  constexpr unsigned AbufSz = 128 * LDA * 2;   // 10240 B
  constexpr unsigned BbufSz = 128 * LDB * 2;   // 10240 B
  __shared__ __align__(16) __bf16 As[2][128 * LDA];
  __shared__ __align__(16) __bf16 Bt[2][128 * LDB];

  const int tid     = threadIdx.x;
  const int rowBase = blockIdx.y * 128;
  const int colBase = blockIdx.x * 128;

  // A loader: 128 rows x 32 halfs; each thread one 32B chunk (two b128 asyncs)
  const int lr = tid >> 1;             // 0..127
  const int lh = (tid & 1) << 4;       // 0 or 16 halfs
  // Wt loader: 128 n-rows x 32 k-halfs; same pattern
  const int bn = tid >> 1;             // 0..127
  const int bk = (tid & 1) << 4;       // 0 or 16 halfs

  const unsigned aOff = (unsigned)(((size_t)(rowBase + lr) * K + lh) * 2);
  const unsigned bOff = (unsigned)(((size_t)(colBase + bn) * K + bk) * 2);
  const unsigned long long aBase = (unsigned long long)A;
  const unsigned long long bBase = (unsigned long long)Wt;

  const unsigned ldsA0 = (unsigned)(uintptr_t)&As[0][0];
  const unsigned ldsB0 = (unsigned)(uintptr_t)&Bt[0][0];
  const unsigned aSlot = (unsigned)((lr * LDA + lh) * 2);
  const unsigned bSlot = (unsigned)((bn * LDB + bk) * 2);

  const int lane = tid & 31;
  const int wave = tid >> 5;
  const int wm   = (wave & 3) * 32;    // wave row offset inside 128
  const int wn   = (wave >> 2) * 64;   // wave col offset inside 128
  const int l16  = lane & 15;
  const int lh2  = lane >> 4;          // 0/1

  // Prime buffer 0 (k0 = 0)
  ASYNC_B128(ldsA0 + aSlot,      aOff,      aBase);
  ASYNC_B128(ldsA0 + aSlot + 16, aOff + 16, aBase);
  ASYNC_B128(ldsB0 + bSlot,      bOff,      bBase);
  ASYNC_B128(ldsB0 + bSlot + 16, bOff + 16, bBase);

  v8f acc[2][4] = {};
  WAIT_ASYNC0();
  __syncthreads();

  for (int k0 = 0; k0 < K; k0 += 32) {
    const int cur = (k0 >> 5) & 1;
    const int nxt = cur ^ 1;
    const bool more = (k0 + 32 < K);
    if (more) {
      const unsigned aO = aOff + (unsigned)((k0 + 32) * 2);
      const unsigned bO = bOff + (unsigned)((k0 + 32) * 2);
      ASYNC_B128(ldsA0 + nxt * AbufSz + aSlot,      aO,      aBase);
      ASYNC_B128(ldsA0 + nxt * AbufSz + aSlot + 16, aO + 16, aBase);
      ASYNC_B128(ldsB0 + nxt * BbufSz + bSlot,      bO,      bBase);
      ASYNC_B128(ldsB0 + nxt * BbufSz + bSlot + 16, bO + 16, bBase);
      if (k0 + 64 < K) {
        __builtin_prefetch((const char*)A  + aOff + (size_t)(k0 + 64) * 2, 0, 1);
        __builtin_prefetch((const char*)Wt + bOff + (size_t)(k0 + 64) * 2, 0, 1);
      }
    }

    const __bf16* Ab = &As[cur][0];
    const __bf16* Bb = &Bt[cur][0];

    // A fragment (ISA 7.12.2, 16-bit A 16x32): lane<16 -> K 0..7 & 16..23; lane>=16 -> +8
    v16bf aF[2];
#pragma unroll
    for (int tm = 0; tm < 2; ++tm) {
      const __bf16* p = Ab + (wm + tm * 16 + l16) * LDA + lh2 * 8;
      v8bf lo = *(const v8bf*)p;
      v8bf hi = *(const v8bf*)(p + 16);
      aF[tm] = __builtin_shufflevector(lo, hi, 0,1,2,3,4,5,6,7,8,9,10,11,12,13,14,15);
    }
    // B fragment (32x16): lanes 0-15 hold K 0..7 & 16..23 of column n, lanes 16-31 hold K +8
    v16bf bF[4];
#pragma unroll
    for (int tn = 0; tn < 4; ++tn) {
      const __bf16* p = Bb + (wn + tn * 16 + l16) * LDB + lh2 * 8;
      v8bf lo = *(const v8bf*)p;
      v8bf hi = *(const v8bf*)(p + 16);
      bF[tn] = __builtin_shufflevector(lo, hi, 0,1,2,3,4,5,6,7,8,9,10,11,12,13,14,15);
    }
#pragma unroll
    for (int tm = 0; tm < 2; ++tm)
#pragma unroll
      for (int tn = 0; tn < 4; ++tn)
        acc[tm][tn] = __builtin_amdgcn_wmma_f32_16x16x32_bf16(
            false, aF[tm], false, bF[tn], (short)0, acc[tm][tn], false, false);

    if (more) {
      WAIT_ASYNC0();     // next tile landed in LDS
      __syncthreads();   // all waves done reading 'cur' and writing 'nxt'
    }
  }

  // Epilogue. C/D layout: VGPR r, lane -> M = r + 8*(lane>=16), N = lane%16
#pragma unroll
  for (int tm = 0; tm < 2; ++tm) {
#pragma unroll
    for (int tn = 0; tn < 4; ++tn) {
      const int col = colBase + wn + tn * 16 + l16;
      const float bv = bias[col];
#pragma unroll
      for (int r = 0; r < 8; ++r) {
        const int row = rowBase + wm + tm * 16 + lh2 * 8 + r;
        float o = acc[tm][tn][r] + bv;
        const size_t idx = (size_t)row * N + col;
        if constexpr (MODE == EP_SILU) {
          o = o / (1.f + __expf(-o));
          ((__bf16*)outv)[idx] = (__bf16)o;
        } else if constexpr (MODE == EP_BIAS) {
          ((__bf16*)outv)[idx] = (__bf16)o;
        } else if constexpr (MODE == EP_GATE) {
          o *= (float)gate[idx];
          ((__bf16*)outv)[idx] = (__bf16)o;
        } else {
          ((float*)outv)[idx] = o;
        }
      }
    }
  }
}

// ---------------- Attention: one (batch, head) per block ----------------
__global__ __launch_bounds__(96)
void attn_kernel(const __bf16* __restrict__ Q, const __bf16* __restrict__ Km,
                 const __bf16* __restrict__ V, __bf16* __restrict__ Y) {
  const int bh = blockIdx.x;
  const int b  = bh >> 4;
  const int h  = bh & 15;
  const int tid = threadIdx.x;

  __shared__ float qs[Tsz * Dd];
  __shared__ float ks[Tsz * Dd];
  __shared__ float vs[Tsz * Dd];
  __shared__ float sc[Tsz * 88];

  const size_t gbase = (size_t)b * Tsz * Csz + (size_t)h * Dd;
  for (int idx = tid; idx < Tsz * Dd; idx += 96) {
    const int r = idx >> 6, d = idx & 63;
    const size_t g = gbase + (size_t)r * Csz + d;
    qs[idx] = (float)Q[g] * 0.125f;   // scale = 1/sqrt(64)
    ks[idx] = (float)Km[g];
    vs[idx] = (float)V[g];
  }
  __syncthreads();

  if (tid < Tsz) {
    const int i  = tid;
    const int im = i % BLKm;
    float qr[Dd];
#pragma unroll
    for (int d = 0; d < Dd; ++d) qr[d] = qs[i * Dd + d];

    float mx = -3.0e38f;
    int jm = 0;
    for (int j = 0; j < Tsz; ++j) {
      if (jm <= im) {                 // mask: (j % 29) <= (i % 29)
        float s = 0.f;
#pragma unroll
        for (int d = 0; d < Dd; ++d) s += qr[d] * ks[j * Dd + d];
        sc[i * 88 + j] = s;
        mx = fmaxf(mx, s);
      }
      if (++jm == BLKm) jm = 0;
    }
    float sum = 0.f;
    jm = 0;
    for (int j = 0; j < Tsz; ++j) {
      if (jm <= im) { float e = __expf(sc[i * 88 + j] - mx); sc[i * 88 + j] = e; sum += e; }
      if (++jm == BLKm) jm = 0;
    }
    const float inv = 1.f / sum;
    float av[Dd];
#pragma unroll
    for (int d = 0; d < Dd; ++d) av[d] = 0.f;
    jm = 0;
    for (int j = 0; j < Tsz; ++j) {
      if (jm <= im) {
        const float w = sc[i * 88 + j] * inv;
#pragma unroll
        for (int d = 0; d < Dd; ++d) av[d] += w * vs[j * Dd + d];
      }
      if (++jm == BLKm) jm = 0;
    }
    __bf16* yo = Y + gbase + (size_t)i * Csz;
#pragma unroll
    for (int d = 0; d < Dd; ++d) yo[d] = (__bf16)av[d];
  }
}

// ---------------- host launch ----------------
extern "C" void kernel_launch(void* const* d_in, const int* in_sizes, int n_in,
                              void* d_out, int out_size, void* d_ws, size_t ws_size,
                              hipStream_t stream) {
  const float* x      = (const float*)d_in[0];
  const float* gamma  = (const float*)d_in[1];
  const float* beta   = (const float*)d_in[2];
  const float* w_act  = (const float*)d_in[3];
  const float* b_act  = (const float*)d_in[4];
  const float* w_in   = (const float*)d_in[5];
  const float* b_in   = (const float*)d_in[6];
  const float* wq     = (const float*)d_in[7];
  const float* bq     = (const float*)d_in[8];
  const float* wk     = (const float*)d_in[9];
  const float* bk     = (const float*)d_in[10];
  const float* w_proj = (const float*)d_in[11];
  const float* b_proj = (const float*)d_in[12];
  const float* w_out  = (const float*)d_in[13];
  const float* b_out  = (const float*)d_in[14];
  float* out = (float*)d_out;

  char* ws = (char*)d_ws;
  size_t off = 0;
  auto take = [&](size_t bytes) -> void* {
    void* p = ws + off;
    off += (bytes + 255) & ~(size_t)255;
    return p;
  };
  const size_t MC = (size_t)Mrows * Csz;
  const size_t CC = (size_t)Csz * Csz;

  __bf16* XN    = (__bf16*)take(MC * 2);
  __bf16* Hb    = (__bf16*)take(MC * 2);
  __bf16* Qb    = (__bf16*)take(MC * 2);
  __bf16* Kb    = (__bf16*)take(MC * 2);
  __bf16* ACT   = (__bf16*)take(MC * 2);
  __bf16* Yb    = (__bf16*)take(MC * 2);
  __bf16* Wact  = (__bf16*)take(CC * 2);   // all weights stored transposed (W^T)
  __bf16* Win   = (__bf16*)take(CC * 2);
  __bf16* Wq    = (__bf16*)take(CC * 2);
  __bf16* Wk    = (__bf16*)take(CC * 2);
  __bf16* Wproj = (__bf16*)take(CC * 2);
  __bf16* Wout  = (__bf16*)take(CC * 2);
  __bf16* Gb = XN;  // XN dead after the first two GEMMs; reuse for gated activations

  const dim3 tg(32, 32);
  transpose_cvt_kernel<<<tg, 256, 0, stream>>>(w_act,  Wact);
  transpose_cvt_kernel<<<tg, 256, 0, stream>>>(w_in,   Win);
  transpose_cvt_kernel<<<tg, 256, 0, stream>>>(wq,     Wq);
  transpose_cvt_kernel<<<tg, 256, 0, stream>>>(wk,     Wk);
  transpose_cvt_kernel<<<tg, 256, 0, stream>>>(w_proj, Wproj);
  transpose_cvt_kernel<<<tg, 256, 0, stream>>>(w_out,  Wout);

  layernorm_kernel<<<Mrows, 256, 0, stream>>>(x, gamma, beta, XN);

  const dim3 gg(Csz / 128, Mrows / 128);  // (8, 174)
  gemm_wmma<EP_SILU><<<gg, 256, 0, stream>>>(XN, Wact, b_act, nullptr, ACT);
  gemm_wmma<EP_SILU><<<gg, 256, 0, stream>>>(XN, Win,  b_in,  nullptr, Hb);
  gemm_wmma<EP_BIAS><<<gg, 256, 0, stream>>>(Hb, Wq, bq, nullptr, Qb);
  gemm_wmma<EP_BIAS><<<gg, 256, 0, stream>>>(Hb, Wk, bk, nullptr, Kb);

  attn_kernel<<<Bsz * Hn, 96, 0, stream>>>(Qb, Kb, Hb, Yb);

  gemm_wmma<EP_GATE><<<gg, 256, 0, stream>>>(Yb, Wproj, b_proj, ACT, Gb);
  gemm_wmma<EP_OUTF><<<gg, 256, 0, stream>>>(Gb, Wout, b_out, nullptr, out);
}
